// EqvPPIFeedForward_51994874085879
// MI455X (gfx1250) — compile-verified, matrix operand order checked by reading, not run on hardware
//
#include <hip/hip_runtime.h>
#include <hip/hip_bf16.h>
#include <math.h>

typedef __attribute__((ext_vector_type(16))) __bf16 v16bf;
typedef __attribute__((ext_vector_type(8)))  float  v8f;

#define NB_    16
#define HID_   64
#define C_     32
#define B_     2
#define NHALF_ 128
#define N_     256

// f32 -> bf16 round-to-nearest-even (bit manipulation; used on one-time paths)
static __device__ __forceinline__ unsigned short f2bf(float f) {
  unsigned int u = __float_as_uint(f);
  unsigned int r = u + 0x7FFFu + ((u >> 16) & 1u);
  return (unsigned short)(r >> 16);
}

// pack two f32 into one dword of bf16 pairs: 2x v_add + 1x v_perm_b32.
// round-half-up (ties differ from RNE; immaterial here).
// D[15:0] = lo[31:16], D[31:16] = hi[31:16] after rounding add.
static __device__ __forceinline__ unsigned int pk_bf16(float lo, float hi) {
  unsigned int a = __float_as_uint(lo) + 0x8000u;
  unsigned int b = __float_as_uint(hi) + 0x8000u;
  return __builtin_amdgcn_perm(b, a, 0x07060302u);
}

union BF16Frag {
  v16bf v;
  unsigned short u[16];
  unsigned int d[8];
};

// ---------------------------------------------------------------------------
// k_init: sx[b,c] = sum_j x[b,j,c]; zero pooled + out_pre accumulators
// ---------------------------------------------------------------------------
__global__ void __launch_bounds__(256) k_init(const float* __restrict__ in1,
                                              const float* __restrict__ in2,
                                              float* __restrict__ sx,
                                              float* __restrict__ pooled,
                                              float* __restrict__ out_pre) {
  int t = threadIdx.x;
  if (t < B_ * C_) {
    int b = t >> 5, c = t & 31;
    float s = 0.f;
    for (int j = 0; j < N_; ++j)
      s += (j < NHALF_) ? in1[(b * NHALF_ + j) * C_ + c]
                        : in2[(b * NHALF_ + (j - NHALF_)) * C_ + c];
    sx[t]     = s;
    pooled[t] = 0.f;
  }
  for (int idx = t; idx < B_ * N_ * C_; idx += 256) out_pre[idx] = 0.f;
}

// ---------------------------------------------------------------------------
// k_y: y[b,j,h,o] = sum_c w_out[h, o*C+c] * x[b,j,c], stored bf16 pre-swizzled
// into the V_WMMA 16-bit A-operand lane layout (rows = o, K = h-chunk):
//   k = e + 8*(e>=8) + 8*kgrp ; lane = (o&15) + 16*kgrp ; group = chunk*2+ot
// one block per (b, j)
// ---------------------------------------------------------------------------
__global__ void __launch_bounds__(64) k_y(const float* __restrict__ in1,
                                          const float* __restrict__ in2,
                                          const float* __restrict__ w_out,
                                          unsigned short* __restrict__ y_sw) {
  __shared__ float xl[C_];
  int bj = blockIdx.x;
  int b = bj / N_, j = bj % N_;
  int t = threadIdx.x;
  if (t < C_) {
    xl[t] = (j < NHALF_) ? in1[(b * NHALF_ + j) * C_ + t]
                         : in2[(b * NHALF_ + (j - NHALF_)) * C_ + t];
  }
  __syncthreads();
  for (int idx = t; idx < HID_ * C_; idx += 64) {
    int h = idx >> 5, o = idx & 31;
    const float* wrow = w_out + (size_t)h * (C_ * C_) + o * C_;
    float s = 0.f;
#pragma unroll
    for (int c = 0; c < C_; ++c) s += wrow[c] * xl[c];
    // inverse of the fragment k-mapping
    int chunk = h >> 5;
    int k     = h & 31;
    int part  = (k >= 16) ? 1 : 0;
    int kk    = k - 16 * part;
    int kgrp  = (kk >= 8) ? 1 : 0;
    int kk2   = kk - 8 * kgrp;
    int e     = part * 8 + kk2;
    int lane  = (o & 15) + 16 * kgrp;
    int ot    = o >> 4;
    size_t off = ((((size_t)bj * 2 + chunk) * 2 + ot) * 32 + lane) * 16 + e;
    y_sw[off] = f2bf(s);
  }
}

// ---------------------------------------------------------------------------
// k_main: BOTH GEMMs on the matrix pipe.
//   GEMM1: hid^T[h,i] = w0^T (const A frags, K padded 16->32) x basis^T (B)
//          + b0 (C bias frags)  -> 4 WMMAs, C layout lane = i-col, rows = h
//   swish + v_perm bf16 pack in registers: GEMM1 C tiles ARE GEMM2's B frags
//   GEMM2: out^T[o,i] += y^T (A frags from pre-swizzled y_sw) x hid^T (B)
//          -> 4 WMMAs accumulating 2 o-tiles
// Block = (b, i_tile, j_segment), 4 waves; global f32 atomics into out_pre.
// ---------------------------------------------------------------------------
__global__ void __launch_bounds__(128) k_main(const float* __restrict__ xyz1,
                                              const float* __restrict__ xyz2,
                                              const float* __restrict__ w0,
                                              const float* __restrict__ b0,
                                              const unsigned short* __restrict__ y_sw,
                                              float* __restrict__ out_pre) {
  __shared__ float w0l[NB_ * HID_];
  __shared__ float b0l[HID_];
  __shared__ float part_[4][16][32];

  int t  = threadIdx.x;
  int wv = t >> 5;        // wave id 0..3
  int L  = t & 31;        // lane
  int blk = blockIdx.x;
  int js = blk & 7;               // j segment (32 j's each)
  int it = (blk >> 3) & 15;       // i tile
  int b  = blk >> 7;              // batch
  int i0 = it << 4;

  for (int idx = t; idx < NB_ * HID_; idx += 128) w0l[idx] = w0[idx];
  if (t < HID_) b0l[t] = b0[t];
  __syncthreads();

  int m    = L & 15;      // i (GEMM1 B-col / GEMM2 D-col); also A-row for w0^T/y^T
  int kgrp = L >> 4;

  // ---- constant fragments: w0^T as 4 A-tiles (h rows, nb K padded to 32),
  //      b0 as 4 C-bias tiles (rows = h, uniform across i columns)
  BF16Frag aw[4];
  v8f cbias[4];
#pragma unroll
  for (int ht = 0; ht < 4; ++ht) {
    int h = ht * 16 + m;
#pragma unroll
    for (int e = 0; e < 16; ++e) {
      int k = e + ((e >= 8) ? 8 : 0) + kgrp * 8;
      aw[ht].u[e] = (k < NB_) ? f2bf(w0l[k * HID_ + h]) : (unsigned short)0;
    }
#pragma unroll
    for (int r = 0; r < 8; ++r) cbias[ht][r] = b0l[ht * 16 + r + 8 * kgrp];
  }

  int i = i0 + m;
  const float* pxi = (i < NHALF_) ? (xyz1 + (size_t)(b * NHALF_ + i) * 3)
                                  : (xyz2 + (size_t)(b * NHALF_ + i - NHALF_) * 3);
  float xi = pxi[0], yi = pxi[1], zi = pxi[2];

  const float step     = 10.0f / 15.0f;  // MAX_RADIUS/(NB-1)
  const float inv_step = 1.5f;

  v8f acc0 = {};   // out^T rows o = 0..15  (lane col = i)
  v8f acc1 = {};   // out^T rows o = 16..31
  const v16bf* yv = (const v16bf*)y_sw;

  int j0 = js * 32;
  for (int j = j0 + wv; j < j0 + 32; j += 4) {
    const float* pxj = (j < NHALF_) ? (xyz1 + (size_t)(b * NHALF_ + j) * 3)
                                    : (xyz2 + (size_t)(b * NHALF_ + j - NHALF_) * 3);
    float dx = xi - pxj[0], dy = yi - pxj[1], dz = zi - pxj[2];
    float d  = sqrtf(dx * dx + dy * dy + dz * dz + 1e-12f);

    // basis^T B fragment: lane needs only 8 bump values (nb = e + 8*kgrp),
    // upper K half (nb >= 16) is the zero padding
    float cb[8];
#pragma unroll
    for (int e = 0; e < 8; ++e) {
      int nb   = e + 8 * kgrp;
      float u  = (d - step * (float)nb) * inv_step;
      float cc = 0.f;
      if (fabsf(u) < 1.f) { float cv = __cosf(1.57079632679f * u); cc = cv * cv; }
      cb[e] = cc;
    }
    BF16Frag bb;
#pragma unroll
    for (int w = 0; w < 4; ++w) bb.d[w] = pk_bf16(cb[2 * w], cb[2 * w + 1]);
#pragma unroll
    for (int w = 4; w < 8; ++w) bb.d[w] = 0u;

    // GEMM1: hid^T tiles (pre-activation), D = w0^T x basis^T + b0
    v8f c0 = __builtin_amdgcn_wmma_f32_16x16x32_bf16(false, aw[0].v, false, bb.v,
                                                     (short)0, cbias[0], false, false);
    v8f c1 = __builtin_amdgcn_wmma_f32_16x16x32_bf16(false, aw[1].v, false, bb.v,
                                                     (short)0, cbias[1], false, false);
    v8f c2 = __builtin_amdgcn_wmma_f32_16x16x32_bf16(false, aw[2].v, false, bb.v,
                                                     (short)0, cbias[2], false, false);
    v8f c3 = __builtin_amdgcn_wmma_f32_16x16x32_bf16(false, aw[3].v, false, bb.v,
                                                     (short)0, cbias[3], false, false);

    // swish, then pack: GEMM1 C rows (r + 8*kgrp) land exactly at GEMM2
    // B-fragment element k = e + 8*(e>=8) + 8*kgrp — same lane, no shuffles
    float s0[8], s1[8], s2[8], s3[8];
#pragma unroll
    for (int r = 0; r < 8; ++r) {
      float v0 = c0[r]; s0[r] = v0 * __builtin_amdgcn_rcpf(1.f + __expf(-v0));
      float v1 = c1[r]; s1[r] = v1 * __builtin_amdgcn_rcpf(1.f + __expf(-v1));
      float v2 = c2[r]; s2[r] = v2 * __builtin_amdgcn_rcpf(1.f + __expf(-v2));
      float v3 = c3[r]; s3[r] = v3 * __builtin_amdgcn_rcpf(1.f + __expf(-v3));
    }
    BF16Frag hb0, hb1;
#pragma unroll
    for (int w = 0; w < 4; ++w) {
      hb0.d[w]     = pk_bf16(s0[2 * w], s0[2 * w + 1]);  // chunk0: h = .. + 8*kgrp
      hb0.d[w + 4] = pk_bf16(s1[2 * w], s1[2 * w + 1]);  // chunk0: h = 16 + ..
      hb1.d[w]     = pk_bf16(s2[2 * w], s2[2 * w + 1]);  // chunk1
      hb1.d[w + 4] = pk_bf16(s3[2 * w], s3[2 * w + 1]);
    }

    // y^T A fragments: contiguous 32B per lane from pre-swizzled y
    size_t jb = ((size_t)(b * N_ + j)) * 4;
    v16bf a00 = yv[(jb + 0) * 32 + L];  // chunk0, o-tile0
    v16bf a01 = yv[(jb + 1) * 32 + L];  // chunk0, o-tile1
    v16bf a10 = yv[(jb + 2) * 32 + L];  // chunk1, o-tile0
    v16bf a11 = yv[(jb + 3) * 32 + L];  // chunk1, o-tile1

    if (j + 4 < j0 + 32)
      __builtin_prefetch((const void*)(y_sw + ((size_t)(b * N_ + j + 4)) * 4 * 512), 0, 0);

    // GEMM2: out^T += y^T x hid^T
    acc0 = __builtin_amdgcn_wmma_f32_16x16x32_bf16(false, a00, false, hb0.v,
                                                   (short)0, acc0, false, false);
    acc1 = __builtin_amdgcn_wmma_f32_16x16x32_bf16(false, a01, false, hb0.v,
                                                   (short)0, acc1, false, false);
    acc0 = __builtin_amdgcn_wmma_f32_16x16x32_bf16(false, a10, false, hb1.v,
                                                   (short)0, acc0, false, false);
    acc1 = __builtin_amdgcn_wmma_f32_16x16x32_bf16(false, a11, false, hb1.v,
                                                   (short)0, acc1, false, false);
  }

  // spill per-wave out^T tiles: D layout lane = i-col (m), rows o = r + 8*kgrp
#pragma unroll
  for (int r = 0; r < 8; ++r) {
    int o = r + 8 * kgrp;
    part_[wv][m][o]      = acc0[r];
    part_[wv][m][16 + o] = acc1[r];
  }
  __syncthreads();

  for (int idx = t; idx < 512; idx += 128) {
    int il = idx >> 5, o = idx & 31;
    float s = part_[0][il][o] + part_[1][il][o] + part_[2][il][o] + part_[3][il][o];
    atomicAdd(&out_pre[((size_t)b * N_ + i0 + il) * C_ + o], s);
  }
}

// ---------------------------------------------------------------------------
// k_norm: add b_out bias term, 1/sqrt(N) scale, per-node RMS norm, mask,
// masked-sum pool into pooled[b,c]. One wave per (b, i_tile); lane = channel.
// ---------------------------------------------------------------------------
__global__ void __launch_bounds__(32) k_norm(const float* __restrict__ out_pre,
                                             const float* __restrict__ b_out,
                                             const float* __restrict__ sx,
                                             const int* __restrict__ mask,
                                             float* __restrict__ pooled) {
  int t  = threadIdx.x;          // channel o
  int b  = blockIdx.x >> 4;
  int i0 = (blockIdx.x & 15) << 4;

  float bias = 0.f;
  for (int c = 0; c < C_; ++c) bias += b_out[t * C_ + c] * sx[b * C_ + c];

  for (int il = 0; il < 16; ++il) {
    int gi = i0 + il;
    float v  = (out_pre[((size_t)b * N_ + gi) * C_ + t] + bias) * (1.0f / 16.0f);
    float ss = v * v;
#pragma unroll
    for (int off = 16; off > 0; off >>= 1) ss += __shfl_xor(ss, off, 32);
    float scale = 1.0f / (sqrtf(ss * (1.0f / 32.0f)) + 1e-6f);
    if (mask[b * N_ + gi] != 0)
      atomicAdd(&pooled[b * C_ + t], v * scale);
  }
}

// ---------------------------------------------------------------------------
// k_final: mean/std (ddof=1) normalize, fc3 + leaky_relu, fc2, sigmoid -> [B]
// ---------------------------------------------------------------------------
__global__ void __launch_bounds__(64) k_final(const float* __restrict__ pooled,
                                              const float* __restrict__ fc3_w,
                                              const float* __restrict__ fc3_b,
                                              const float* __restrict__ fc2_w,
                                              const float* __restrict__ fc2_b,
                                              float* __restrict__ out) {
  __shared__ float pl[64], zl[64], z2[64], mv[2], sv[2];
  int t = threadIdx.x;
  pl[t] = pooled[t];
  __syncthreads();
  if (t < 2) {
    float s = 0.f;
    for (int c = 0; c < 32; ++c) s += pl[t * 32 + c];
    float mean = s / 32.f;
    float v = 0.f;
    for (int c = 0; c < 32; ++c) { float d = pl[t * 32 + c] - mean; v += d * d; }
    mv[t] = mean;
    sv[t] = sqrtf(v / 31.f);     // ddof = 1
  }
  __syncthreads();
  int b = t >> 5, o = t & 31;
  zl[t] = (pl[t] - mv[b]) / (sv[b] + 1e-6f);
  __syncthreads();
  {
    float a = fc3_b[o];
    for (int c = 0; c < 32; ++c) a += zl[b * 32 + c] * fc3_w[c * 32 + o];
    z2[t] = (a > 0.f) ? a : 0.01f * a;   // leaky_relu
  }
  __syncthreads();
  if (t < 2) {
    float s = fc2_b[0];
    for (int o2 = 0; o2 < 32; ++o2) s += z2[t * 32 + o2] * fc2_w[o2];
    out[t] = 1.0f / (1.0f + __expf(-s));
  }
}

// ---------------------------------------------------------------------------
extern "C" void kernel_launch(void* const* d_in, const int* in_sizes, int n_in,
                              void* d_out, int out_size, void* d_ws, size_t ws_size,
                              hipStream_t stream) {
  const float* in1   = (const float*)d_in[0];
  const float* in2   = (const float*)d_in[1];
  const float* xyz1  = (const float*)d_in[2];
  const float* xyz2  = (const float*)d_in[3];
  const int*   mask  = (const int*)d_in[4];
  const float* w0    = (const float*)d_in[5];
  const float* b0    = (const float*)d_in[6];
  const float* w_out = (const float*)d_in[7];
  const float* b_out = (const float*)d_in[8];
  const float* fc3_w = (const float*)d_in[9];
  const float* fc3_b = (const float*)d_in[10];
  const float* fc2_w = (const float*)d_in[11];
  const float* fc2_b = (const float*)d_in[12];

  // workspace layout: [0,2MB) y_sw bf16 | sx 64f | pooled 64f | out_pre 16384f
  unsigned short* y_sw = (unsigned short*)d_ws;
  float* sx      = (float*)((char*)d_ws + (2u << 20));
  float* pooled  = sx + 64;
  float* out_pre = pooled + 64;
  float* out     = (float*)d_out;

  k_init <<<1,            256, 0, stream>>>(in1, in2, sx, pooled, out_pre);
  k_y    <<<B_ * N_,       64, 0, stream>>>(in1, in2, w_out, y_sw);
  k_main <<<B_ * 16 * 8,  128, 0, stream>>>(xyz1, xyz2, w0, b0, y_sw, out_pre);
  k_norm <<<B_ * 16,       32, 0, stream>>>(out_pre, b_out, sx, mask, pooled);
  k_final<<<1,             64, 0, stream>>>(pooled, fc3_w, fc3_b, fc2_w, fc2_b, out);
}
